// ImprovedGraphAttentionLayer_43688407335636
// MI455X (gfx1250) — matrix operand order repaired and checked by reading, not compiled, not run on hardware
//
#include <hip/hip_runtime.h>
#include <hip/hip_bf16.h>

typedef __attribute__((ext_vector_type(16))) _Float16 v16h;
typedef __attribute__((ext_vector_type(8)))  float    v8f;

#define BB  2
#define NN  384
#define HID 128
#define NH  8
#define HD  16
#define ED  128
#define TP  128          // pairs per block in edge kernel (never crosses b or i boundary)
#define EDP 136          // padded LDS stride (halves): 16B-aligned chunks, conflict-free

// ---------------- cross-lane helpers (VALU DPP + one ds_swizzle) ----------------
template <int CTRL>
__device__ __forceinline__ float dpp_mov(float v) {
  return __int_as_float(
      __builtin_amdgcn_update_dpp(0, __float_as_int(v), CTRL, 0xf, 0xf, true));
}
__device__ __forceinline__ float row16_sum(float v) {
  v += dpp_mov<0xB1>(v);   // quad_perm [1,0,3,2]
  v += dpp_mov<0x4E>(v);   // quad_perm [2,3,0,1]
  v += dpp_mov<0x141>(v);  // row_half_mirror
  v += dpp_mov<0x140>(v);  // row_mirror
  return v;
}
__device__ __forceinline__ float row16_max(float v) {
  v = fmaxf(v, dpp_mov<0xB1>(v));
  v = fmaxf(v, dpp_mov<0x4E>(v));
  v = fmaxf(v, dpp_mov<0x141>(v));
  v = fmaxf(v, dpp_mov<0x140>(v));
  return v;
}
__device__ __forceinline__ float swz16(float v) {  // SWAPX16: xor=0x10,and=0x1f
  return __int_as_float(__builtin_amdgcn_ds_swizzle(__float_as_int(v), 0x401F));
}
__device__ __forceinline__ float wave_sum(float v) { v = row16_sum(v); return v + swz16(v); }
__device__ __forceinline__ float wave_max(float v) { v = row16_max(v); return fmaxf(v, swz16(v)); }

// ---------------- weights f32 -> f16 pre-permuted into B-fragment order ------
__global__ __launch_bounds__(256) void k_cvtw(const float* __restrict__ We,
                                              const float* __restrict__ Wg,
                                              _Float16* __restrict__ WHB) {
  int idx = blockIdx.x * 256 + threadIdx.x;
  if (idx >= 2 * HID * HID) return;
  int e    = idx & 15;
  int lane = (idx >> 4) & 31;
  int ks   = (idx >> 9) & 3;
  int m    = (idx >> 11) & 1;
  int h    = idx >> 12;
  int k = ks * 32 + (lane >> 4) * 8 + (e & 7) + (e >> 3) * 16;
  int n = h * HD + (lane & 15);
  WHB[idx] = (_Float16)((m == 0 ? We : Wg)[k * HID + n]);
}

// ---------------- node projection ----------------
__global__ __launch_bounds__(128) void k_nodeproj(const float* __restrict__ node,
                                                  const float* __restrict__ W,
                                                  const float* __restrict__ bias,
                                                  float* __restrict__ h0) {
  __shared__ float sx[HID];
  int row = blockIdx.x, t = threadIdx.x;
  sx[t] = node[(size_t)row * HID + t];
  __syncthreads();
  float acc = bias[t];
  #pragma unroll 4
  for (int k = 0; k < HID; ++k) acc += sx[k] * W[k * HID + t];
  h0[(size_t)row * HID + t] = acc;
}

// ---------------- QKV projections ----------------
__global__ __launch_bounds__(128) void k_qkv(const float* __restrict__ h0,
    const float* __restrict__ Wq, const float* __restrict__ bq,
    const float* __restrict__ Wk, const float* __restrict__ bk,
    const float* __restrict__ Wv, const float* __restrict__ bv,
    float* __restrict__ Q, float* __restrict__ K, float* __restrict__ V) {
  __shared__ float sx[HID];
  int row = blockIdx.x, t = threadIdx.x;
  sx[t] = h0[(size_t)row * HID + t];
  __syncthreads();
  float aq = bq[t], ak = bk[t], av = bv[t];
  #pragma unroll 4
  for (int k = 0; k < HID; ++k) {
    float x = sx[k];
    aq += x * Wq[k * HID + t];
    ak += x * Wk[k * HID + t];
    av += x * Wv[k * HID + t];
  }
  Q[(size_t)row * HID + t] = aq;
  K[(size_t)row * HID + t] = ak;
  V[(size_t)row * HID + t] = av;
}

// ---------------- fused edge GEMM + gate + QK scores + mask ----------------
__global__ __launch_bounds__(256) void k_edge_scores(
    const float* __restrict__ edge, const _Float16* __restrict__ WHB,
    const float* __restrict__ bedge, const float* __restrict__ bgate,
    const float* __restrict__ Q, const float* __restrict__ Kmat,
    const int* __restrict__ adj, float* __restrict__ scores) {
  __shared__ _Float16 sA[TP * EDP];   // ~34 KB pair tile, f16

  const size_t p0 = (size_t)blockIdx.x * TP;
  const int b  = (int)(p0 / (size_t)(NN * NN));
  const int r2 = (int)(p0 - (size_t)b * NN * NN);
  const int i  = r2 / NN;
  const int j0 = r2 - i * NN;
  const int t  = threadIdx.x;

  // vectorized A-tile staging: float4 global loads, packed b64 LDS stores
  const float4* e4 = (const float4*)(edge + p0 * ED);
  for (int p = t; p < TP * ED / 4; p += 256) {
    float4 x = e4[p];
    int r = p >> 5;            // 32 float4 per row
    int c4 = (p & 31) * 4;
    union { _Float16 hh[4]; uint2 u; } pk;
    pk.hh[0] = (_Float16)x.x; pk.hh[1] = (_Float16)x.y;
    pk.hh[2] = (_Float16)x.z; pk.hh[3] = (_Float16)x.w;
    *(uint2*)&sA[r * EDP + c4] = pk.u;
  }
  __syncthreads();

  const int h = t >> 5, lane = t & 31;
  const int nI = lane & 15, kq = lane >> 4;

  v16h bfe[4], bfg[4];
  #pragma unroll
  for (int ks = 0; ks < 4; ++ks) {
    bfe[ks] = *(const v16h*)(WHB + ((((h * 2 + 0) * 4 + ks) * 32 + lane) << 4));
    bfg[ks] = *(const v16h*)(WHB + ((((h * 2 + 1) * 4 + ks) * 32 + lane) << 4));
  }
  const float bbE = bedge[h * HD + nI];
  const float bbG = bgate[h * HD + nI];
  const float qv  = Q[((size_t)b * NN + i) * HID + h * HD + nI];
  // per-sub-tile K base pointer; 8 loads at constant 512B-stride immediate offsets
  const float* kp = Kmat + ((size_t)b * NN + j0 + kq * 8) * HID + h * HD + nI;
  const int* adjp = adj + p0;
  float* scout = scores + (((size_t)b * NH + h) * NN + i) * NN + j0;

  for (int m0 = 0; m0 < TP; m0 += 16) {
    float kvv[8];
    #pragma unroll
    for (int r = 0; r < 8; ++r) kvv[r] = kp[r * HID];   // immediate offsets
    kp += 16 * HID;                                     // one 64-bit add per sub-tile

    v8f ae = {}, ag = {};
    const int mrow = m0 + nI;
    #pragma unroll
    for (int ks = 0; ks < 4; ++ks) {
      alignas(32) _Float16 ta[16];
      #pragma unroll
      for (int e = 0; e < 8; ++e) ta[e]     = sA[mrow * EDP + ks * 32 + kq * 8 + e];
      #pragma unroll
      for (int e = 0; e < 8; ++e) ta[8 + e] = sA[mrow * EDP + ks * 32 + 16 + kq * 8 + e];
      v16h a = *(const v16h*)ta;
      ae = __builtin_amdgcn_wmma_f32_16x16x32_f16(false, a, false, bfe[ks], (short)0, ae, false, false);
      ag = __builtin_amdgcn_wmma_f32_16x16x32_f16(false, a, false, bfg[ks], (short)0, ag, false, false);
    }
    // gated reduction + lane-parallel Q·K dot; keep row r's sum on lane nI&7==r
    float keep = 0.f;
    #pragma unroll
    for (int r = 0; r < 8; ++r) {
      float g   = ag[r] + bbG;
      float sgm = __builtin_amdgcn_rcpf(1.0f + __expf(-g));
      float pv  = (ae[r] + bbE) * sgm + qv * kvv[r] * 0.25f;
      pv = row16_sum(pv);
      if ((nI & 7) == r) keep = pv;
    }
    // single coalesced 16-lane store per sub-tile
    const int mst = m0 + kq * 8 + (nI & 7);
    if (nI < 8) {
      float sc = keep;
      if (adjp[mst] == 0) sc = -1e9f;
      scout[mst] = sc;
    }
  }
}

// ---------------- softmax: one wave per row (384 = 32 lanes x 3 float4) --------
__global__ __launch_bounds__(256) void k_softmax(float* __restrict__ scores) {
  size_t row = (size_t)blockIdx.x * 8 + (threadIdx.x >> 5);
  int lane = threadIdx.x & 31;
  float* p = scores + row * NN + lane * 4;
  float4 x0 = *(float4*)(p);
  float4 x1 = *(float4*)(p + 128);
  float4 x2 = *(float4*)(p + 256);
  float mx = fmaxf(fmaxf(fmaxf(x0.x, x0.y), fmaxf(x0.z, x0.w)),
             fmaxf(fmaxf(fmaxf(x1.x, x1.y), fmaxf(x1.z, x1.w)),
                   fmaxf(fmaxf(x2.x, x2.y), fmaxf(x2.z, x2.w))));
  mx = wave_max(mx);
  x0.x = __expf(x0.x - mx); x0.y = __expf(x0.y - mx); x0.z = __expf(x0.z - mx); x0.w = __expf(x0.w - mx);
  x1.x = __expf(x1.x - mx); x1.y = __expf(x1.y - mx); x1.z = __expf(x1.z - mx); x1.w = __expf(x1.w - mx);
  x2.x = __expf(x2.x - mx); x2.y = __expf(x2.y - mx); x2.z = __expf(x2.z - mx); x2.w = __expf(x2.w - mx);
  float s = ((x0.x + x0.y) + (x0.z + x0.w)) + ((x1.x + x1.y) + (x1.z + x1.w)) +
            ((x2.x + x2.y) + (x2.z + x2.w));
  float inv = __builtin_amdgcn_rcpf(wave_sum(s));
  x0.x *= inv; x0.y *= inv; x0.z *= inv; x0.w *= inv;
  x1.x *= inv; x1.y *= inv; x1.z *= inv; x1.w *= inv;
  x2.x *= inv; x2.y *= inv; x2.z *= inv; x2.w *= inv;
  *(float4*)(p) = x0; *(float4*)(p + 128) = x1; *(float4*)(p + 256) = x2;
}

// ---------------- attn @ V per head (WMMA) ----------------
__global__ __launch_bounds__(256) void k_attnv(const float* __restrict__ attn,
                                               const float* __restrict__ V,
                                               float* __restrict__ aout) {
  int blk = blockIdx.x;
  int b = blk / (NH * 3);
  int rem = blk - b * (NH * 3);
  int h = rem / 3;
  int grp = rem - h * 3;
  int t = threadIdx.x;
  int w = t >> 5, lane = t & 31;
  int nI = lane & 15, kq = lane >> 4;
  int i0 = (grp * 8 + w) * 16;
  // strength-reduced base pointers, advanced per K-step
  const float* ar = attn + ((size_t)b * NH + h) * NN * NN + (size_t)(i0 + nI) * NN + kq * 16;
  const float* vp = V + ((size_t)b * NN + kq * 16) * HID + h * HD + nI;

  v8f acc = {};
  for (int ks = 0; ks < NN / 32; ++ks) {
    alignas(32) _Float16 ta[16];
    #pragma unroll
    for (int e = 0; e < 8; ++e) ta[e]     = (_Float16)ar[e];       // immediate offsets
    #pragma unroll
    for (int e = 0; e < 8; ++e) ta[8 + e] = (_Float16)ar[8 + e];
    v16h a = *(const v16h*)ta;
    alignas(32) _Float16 tb[16];
    #pragma unroll
    for (int e = 0; e < 16; ++e) tb[e] = (_Float16)vp[e * HID];    // immediate offsets
    v16h bf = *(const v16h*)tb;
    acc = __builtin_amdgcn_wmma_f32_16x16x32_f16(false, a, false, bf, (short)0, acc, false, false);
    ar += 32;
    vp += 32 * HID;
  }
  #pragma unroll
  for (int r = 0; r < 8; ++r) {
    int m = r + kq * 8;
    aout[((size_t)b * NN + i0 + m) * HID + h * HD + nI] = acc[r];
  }
}

// ---------------- out projection + residual + LayerNorm ----------------
__global__ __launch_bounds__(128) void k_outln(const float* __restrict__ aout,
    const float* __restrict__ Wout, const float* __restrict__ bout,
    const float* __restrict__ resid, const float* __restrict__ g1,
    const float* __restrict__ be1, float* __restrict__ hfin,
    float* __restrict__ out_h) {
  __shared__ float sx[HID];
  __shared__ float red[16];
  int row = blockIdx.x, t = threadIdx.x;
  sx[t] = aout[(size_t)row * HID + t];
  __syncthreads();
  float acc = bout[t];
  #pragma unroll 4
  for (int k = 0; k < HID; ++k) acc += sx[k] * Wout[k * HID + t];
  float v = resid[(size_t)row * HID + t] + acc;
  float s  = row16_sum(v);
  float s2 = row16_sum(v * v);
  if ((t & 15) == 0) { red[t >> 4] = s; red[8 + (t >> 4)] = s2; }
  __syncthreads();
  float sum   = (red[0] + red[1]) + (red[2] + red[3]) + (red[4] + red[5]) + (red[6] + red[7]);
  float sumsq = (red[8] + red[9]) + (red[10] + red[11]) + (red[12] + red[13]) + (red[14] + red[15]);
  float mean = sum * (1.0f / HID);
  float var  = sumsq * (1.0f / HID) - mean * mean;
  float o = (v - mean) * __builtin_amdgcn_rsqf(var + 1e-5f) * g1[t] + be1[t];
  hfin[(size_t)row * HID + t] = o;
  out_h[(size_t)row * HID + t] = o;
}

// ---------------- hp = h @ W_eout ----------------
__global__ __launch_bounds__(128) void k_hp(const float* __restrict__ hfin,
                                            const float* __restrict__ Weout,
                                            float* __restrict__ hp) {
  __shared__ float sx[HID];
  int row = blockIdx.x, t = threadIdx.x;
  sx[t] = hfin[(size_t)row * HID + t];
  __syncthreads();
  float acc = 0.f;
  #pragma unroll 4
  for (int k = 0; k < HID; ++k) acc += sx[k] * Weout[k * ED + t];
  hp[(size_t)row * ED + t] = acc;
}

// ---------------- edge output: one wave per pair, float4 streaming, no LDS -----
__global__ __launch_bounds__(256) void k_edgeout(const float* __restrict__ edge,
    const float* __restrict__ attn, const float* __restrict__ hp,
    const float* __restrict__ beout, const float* __restrict__ g2,
    const float* __restrict__ be2, float* __restrict__ out_e) {
  size_t pair = (size_t)blockIdx.x * 8 + (threadIdx.x >> 5);
  int lane = threadIdx.x & 31;
  int b = (int)(pair / (size_t)(NN * NN));
  int r2 = (int)(pair - (size_t)b * NN * NN);
  int i = r2 / NN;
  int j = r2 - i * NN;
  // mean over heads: lanes 0..7 load, wave-reduce (others contribute 0)
  float a = (lane < NH) ? attn[(((size_t)b * NH + lane) * NN + i) * NN + j] : 0.f;
  float am = wave_sum(a) * (0.5f / NH);   // folds the 0.5 pair factor
  int c = lane * 4;
  float4 ev = *(const float4*)(edge + pair * ED + c);
  float4 pi = *(const float4*)(hp + ((size_t)b * NN + i) * ED + c);
  float4 pj = *(const float4*)(hp + ((size_t)b * NN + j) * ED + c);
  float4 bo = *(const float4*)(beout + c);
  float v0 = ev.x + am * (pi.x + pj.x) + bo.x;
  float v1 = ev.y + am * (pi.y + pj.y) + bo.y;
  float v2 = ev.z + am * (pi.z + pj.z) + bo.z;
  float v3 = ev.w + am * (pi.w + pj.w) + bo.w;
  float s  = wave_sum((v0 + v1) + (v2 + v3));
  float s2 = wave_sum((v0 * v0 + v1 * v1) + (v2 * v2 + v3 * v3));
  float mean = s * (1.0f / ED);
  float var  = s2 * (1.0f / ED) - mean * mean;
  float rs = __builtin_amdgcn_rsqf(var + 1e-5f);
  float4 gg = *(const float4*)(g2 + c);
  float4 bb = *(const float4*)(be2 + c);
  float4 o;
  o.x = (v0 - mean) * rs * gg.x + bb.x;
  o.y = (v1 - mean) * rs * gg.y + bb.y;
  o.z = (v2 - mean) * rs * gg.z + bb.z;
  o.w = (v3 - mean) * rs * gg.w + bb.w;
  *(float4*)(out_e + pair * ED + c) = o;
}

extern "C" void kernel_launch(void* const* d_in, const int* in_sizes, int n_in,
                              void* d_out, int out_size, void* d_ws, size_t ws_size,
                              hipStream_t stream) {
  const float* node  = (const float*)d_in[0];
  const float* edge  = (const float*)d_in[1];
  const int*   adj   = (const int*)d_in[2];
  const float* Wnode = (const float*)d_in[3];
  const float* bnode = (const float*)d_in[4];
  const float* Wq = (const float*)d_in[5];  const float* bq = (const float*)d_in[6];
  const float* Wk = (const float*)d_in[7];  const float* bk = (const float*)d_in[8];
  const float* Wv = (const float*)d_in[9];  const float* bv = (const float*)d_in[10];
  const float* Wedge = (const float*)d_in[11]; const float* bedge = (const float*)d_in[12];
  const float* Wgate = (const float*)d_in[13]; const float* bgate = (const float*)d_in[14];
  const float* Wout  = (const float*)d_in[15]; const float* bout  = (const float*)d_in[16];
  const float* Weout = (const float*)d_in[17]; const float* beout = (const float*)d_in[18];
  const float* g1 = (const float*)d_in[19]; const float* be1 = (const float*)d_in[20];
  const float* g2 = (const float*)d_in[21]; const float* be2 = (const float*)d_in[22];
  (void)in_sizes; (void)n_in; (void)out_size; (void)ws_size;

  float* ws = (float*)d_ws;
  const size_t R = (size_t)BB * NN * HID;          // 98304
  float* h0     = ws;
  float* Q      = h0 + R;
  float* K      = Q + R;
  float* V      = K + R;
  float* scores = V + R;                            // B*NH*N*N = 2359296
  float* aout   = scores + (size_t)BB * NH * NN * NN;
  float* hfin   = aout + R;
  float* hp     = hfin + R;
  _Float16* WHB = (_Float16*)(hp + R);              // 2*128*128 halves, B-fragment order

  float* out_h = (float*)d_out;
  float* out_e = out_h + R;

  const int rows = BB * NN;
  k_cvtw    <<<(2 * HID * HID + 255) / 256, 256, 0, stream>>>(Wedge, Wgate, WHB);
  k_nodeproj<<<rows, 128, 0, stream>>>(node, Wnode, bnode, h0);
  k_qkv     <<<rows, 128, 0, stream>>>(h0, Wq, bq, Wk, bk, Wv, bv, Q, K, V);
  k_edge_scores<<<(BB * NN * NN) / TP, 256, 0, stream>>>(edge, WHB, bedge, bgate, Q, K, adj, scores);
  k_softmax <<<(BB * NH * NN) / 8, 256, 0, stream>>>(scores);
  k_attnv   <<<BB * NH * 3, 256, 0, stream>>>(scores, V, aout);
  k_outln   <<<rows, 128, 0, stream>>>(aout, Wout, bout, h0, g1, be1, hfin, out_h);
  k_hp      <<<rows, 128, 0, stream>>>(hfin, Weout, hp);
  k_edgeout <<<(BB * NN * NN) / 8, 256, 0, stream>>>(edge, scores, hp, beout, g2, be2, out_e);
}
